// GCN_6038724018181
// MI455X (gfx1250) — matrix operand order repaired and checked by reading, not compiled, not run on hardware
//
#include <hip/hip_runtime.h>
#include <hip/hip_bf16.h>

typedef __attribute__((ext_vector_type(16))) __bf16 v16bf;
typedef __attribute__((ext_vector_type(8)))  float  v8f;
typedef int v4i __attribute__((vector_size(16)));   // matches builtin prototype

#define D_FEAT 64

#if defined(__has_builtin)
#if __has_builtin(__builtin_amdgcn_global_load_async_to_lds_b128) && \
    __has_builtin(__builtin_amdgcn_s_wait_asynccnt)
#define GCN_ASYNC_STAGE 1
#endif
#endif

#ifdef GCN_ASYNC_STAGE
// flat->AS1/AS3 via integer cast; for LDS the low 32 bits of the flat address
// are the LDS byte offset.
#define GCN_AS1(p) ((__attribute__((address_space(1))) v4i*)(unsigned long long)(const void*)(p))
#define GCN_AS3(p) ((__attribute__((address_space(3))) v4i*)(unsigned int)(unsigned long long)(void*)(p))
#endif

// ---------------------------------------------------------------------------
// zero-fill (graph-capture safe replacement for hipMemsetAsync)
// ---------------------------------------------------------------------------
__global__ __launch_bounds__(256) void gcn_zero(float4* __restrict__ p, long long nvec) {
    long long i = (long long)blockIdx.x * blockDim.x + threadIdx.x;
    if (i < nvec) p[i] = make_float4(0.f, 0.f, 0.f, 0.f);
}

// ---------------------------------------------------------------------------
// SPMM: out[dst] += d[src]*d[dst]*x[src]   (one wave32 per edge, float2/lane)
// ---------------------------------------------------------------------------
__global__ __launch_bounds__(256) void gcn_spmm(const float* __restrict__ X,
                                                const int*   __restrict__ edges,
                                                const float* __restrict__ dinv,
                                                float* __restrict__ out,
                                                long long E) {
    long long tid  = (long long)blockIdx.x * blockDim.x + threadIdx.x;
    long long e    = tid >> 5;
    int       lane = (int)(tid & 31);
    if (e >= E) return;
    int src = edges[e];
    int dst = edges[E + e];
    float coef = dinv[src] * dinv[dst];
    float2 v = ((const float2*)(X + (size_t)src * D_FEAT))[lane];
    float* o = out + (size_t)dst * D_FEAT + 2 * lane;
    atomicAdd(o,     v.x * coef);
    atomicAdd(o + 1, v.y * coef);
}

// ---------------------------------------------------------------------------
// GEMM: out = (relu?) (A[nrows x 64] @ W[64 x 64])  via v_wmma_f32_16x16x32_bf16
// One wave computes a 16(M) x 64(N) strip: 4 n-tiles x 2 chained K=32 WMMAs.
// W is staged to LDS (async if available), then pre-swizzled once per block
// into fragment-major bf16 layout so each lane reads its whole B fragment
// with wide ds_load_b128s.
// ---------------------------------------------------------------------------
__global__ __launch_bounds__(256) void gcn_gemm(const float* __restrict__ A,
                                                const float* __restrict__ W,
                                                float* __restrict__ out,
                                                int nrows, int do_relu) {
    __shared__ float  sWf[D_FEAT * D_FEAT];                 // raw W (16 KB)
    __shared__ __align__(32) __bf16 sWb[8 * 32 * 16];       // fragments (8 KB)

    // ---- stage raw W into LDS --------------------------------------------
    {
        int off = threadIdx.x * 4;                          // 4 floats / op
#ifdef GCN_ASYNC_STAGE
#pragma unroll
        for (int i = 0; i < 4; ++i)
            __builtin_amdgcn_global_load_async_to_lds_b128(
                GCN_AS1(W + off + i * 1024), GCN_AS3(sWf + off + i * 1024), 0, 0);
        __builtin_amdgcn_s_wait_asynccnt(0);
#else
#pragma unroll
        for (int i = 0; i < 4; ++i)
            *(float4*)(sWf + off + i * 1024) = *(const float4*)(W + off + i * 1024);
#endif
    }
    __syncthreads();

    // ---- pre-swizzle into B-fragment layout: [nt*2+kt][lane][e] ----------
    // lane n holds column nt*16+(n&15), K = kt*32 + (n>>4)*8 + (e&7) + (e>>3)*16
    for (int j = threadIdx.x; j < 8 * 32 * 16; j += 256) {
        int e    = j & 15;
        int ln   = (j >> 4) & 31;
        int frag = j >> 9;
        int nt = frag >> 1, kt = frag & 1;
        int col = nt * 16 + (ln & 15);
        int k   = kt * 32 + (ln >> 4) * 8 + (e & 7) + (e >> 3) * 16;
        sWb[j] = (__bf16)sWf[k * D_FEAT + col];
    }
    __syncthreads();

    int wave = threadIdx.x >> 5;        // 8 waves / block
    int lane = threadIdx.x & 31;
    int tile = blockIdx.x * 8 + wave;   // 16-row tile index (wave-uniform)
    int row0 = tile * 16;
    if (row0 >= nrows) return;          // uniform exit -> EXEC all-1s for WMMA

    int rlane = lane & 15;
    int hi    = lane >> 4;              // 0: lanes 0-15, 1: lanes 16-31
    const float* arow = A + (size_t)(row0 + rlane) * D_FEAT;

    // A fragments (16x32 bf16 each): lane holds K in [kb,kb+8) and
    // [kb+16,kb+24), kb = kt*32 + hi*8.
    v16bf afrag[2];
#pragma unroll
    for (int kt = 0; kt < 2; ++kt) {
        int kb = kt * 32 + hi * 8;
        float4 p0 = *(const float4*)(arow + kb);
        float4 p1 = *(const float4*)(arow + kb + 4);
        float4 p2 = *(const float4*)(arow + kb + 16);
        float4 p3 = *(const float4*)(arow + kb + 20);
        v16bf a;
        a[0]  = (__bf16)p0.x; a[1]  = (__bf16)p0.y; a[2]  = (__bf16)p0.z; a[3]  = (__bf16)p0.w;
        a[4]  = (__bf16)p1.x; a[5]  = (__bf16)p1.y; a[6]  = (__bf16)p1.z; a[7]  = (__bf16)p1.w;
        a[8]  = (__bf16)p2.x; a[9]  = (__bf16)p2.y; a[10] = (__bf16)p2.z; a[11] = (__bf16)p2.w;
        a[12] = (__bf16)p3.x; a[13] = (__bf16)p3.y; a[14] = (__bf16)p3.z; a[15] = (__bf16)p3.w;
        afrag[kt] = a;
    }

#pragma unroll
    for (int nt = 0; nt < 4; ++nt) {
        v8f c = {};
#pragma unroll
        for (int kt = 0; kt < 2; ++kt) {
            // whole 32-byte B fragment for this lane: wide LDS load
            v16bf b = *(const v16bf*)(sWb + (((nt * 2 + kt) * 32 + lane) << 4));
            c = __builtin_amdgcn_wmma_f32_16x16x32_bf16(
                    false, afrag[kt], false, b, (short)0, c, false, false);
        }
        // C/D layout: VGPR r -> M = r + 8*hi, N = rlane
        float* ob = out + (size_t)row0 * D_FEAT + nt * 16;
#pragma unroll
        for (int r = 0; r < 8; ++r) {
            float v = c[r];
            if (do_relu) v = v > 0.f ? v : 0.f;
            ob[(size_t)(r + 8 * hi) * D_FEAT + rlane] = v;
        }
    }
}

// ---------------------------------------------------------------------------
extern "C" void kernel_launch(void* const* d_in, const int* in_sizes, int n_in,
                              void* d_out, int out_size, void* d_ws, size_t ws_size,
                              hipStream_t stream) {
    const float* x     = (const float*)d_in[0];
    const int*   edges = (const int*)  d_in[1];
    const float* dinv  = (const float*)d_in[2];
    const float* W1    = (const float*)d_in[4];
    const float* W2    = (const float*)d_in[5];
    float*       outp  = (float*)d_out;

    int       n = in_sizes[0] / D_FEAT;        // 50000 nodes
    long long E = in_sizes[1] / 2;             // 850000 edges

    float* buf0 = (float*)d_ws;                      // aggregation buffer
    float* buf1 = buf0 + (size_t)n * D_FEAT;         // hidden activations

    long long nvec = (long long)n * D_FEAT / 4;
    int zblocks    = (int)((nvec + 255) / 256);
    long long sthr = E * 32;
    int sblocks    = (int)((sthr + 255) / 256);
    int gblocks    = (n + 127) / 128;                // 8 waves x 16 rows

    // layer 1: agg = A_norm @ X ; h = relu(agg @ W1)
    gcn_zero<<<zblocks, 256, 0, stream>>>((float4*)buf0, nvec);
    gcn_spmm<<<sblocks, 256, 0, stream>>>(x, edges, dinv, buf0, E);
    gcn_gemm<<<gblocks, 256, 0, stream>>>(buf0, W1, buf1, n, 1);

    // layer 2: agg = A_norm @ h ; out = agg @ W2
    gcn_zero<<<zblocks, 256, 0, stream>>>((float4*)buf0, nvec);
    gcn_spmm<<<sblocks, 256, 0, stream>>>(buf1, edges, dinv, buf0, E);
    gcn_gemm<<<gblocks, 256, 0, stream>>>(buf0, W2, outp, n, 0);
}
